// DCRNNModel_classification_44074954392027
// MI455X (gfx1250) — matrix operand order repaired and checked by reading, not compiled, not run on hardware
//
#include <hip/hip_runtime.h>
#include <hip/hip_bf16.h>

typedef __attribute__((ext_vector_type(16))) __bf16 v16bf;
typedef __attribute__((ext_vector_type(8)))  float  v8f;
typedef unsigned int u32x4 __attribute__((ext_vector_type(4)));
typedef int          i32x8 __attribute__((ext_vector_type(8)));
typedef int          i32x4 __attribute__((ext_vector_type(4)));

#if defined(__has_builtin)
#if __has_builtin(__builtin_amdgcn_tensor_load_to_lds) && __has_builtin(__builtin_amdgcn_s_wait_tensorcnt)
#define HAVE_TDM 1
#endif
#endif
#ifndef HAVE_TDM
#define HAVE_TDM 0
#endif

union Frag16 {
    v16bf v;
    unsigned short u[16];
    uint4 q[2];
};
union Pack8 {
    uint4 q;
    unsigned short u[8];
};

__device__ __forceinline__ unsigned short f2bf(float f) {
    unsigned int u = __float_as_uint(f);
    unsigned int r = (u + 0x7FFFu + ((u >> 16) & 1u)) >> 16;
    return (unsigned short)r;
}
__device__ __forceinline__ float bf2f(unsigned short h) {
    return __uint_as_float(((unsigned int)h) << 16);
}
__device__ __forceinline__ float uhash(unsigned a, unsigned b, unsigned c) {
    unsigned h = a * 0x9E3779B1u ^ (b + 0x85EBCA77u) * 0xC2B2AE3Du ^ (c + 0x27D4EB2Fu) * 0x165667B1u;
    h ^= h >> 16; h *= 0x7FEB352Du; h ^= h >> 15; h *= 0x846CA68Bu; h ^= h >> 16;
    float u = ((float)h + 0.5f) * (1.0f / 4294967296.0f);
    return fminf(fmaxf(u, 1e-12f), 1.0f - 1e-7f);
}

__device__ __forceinline__ v8f wmma_bf16(v16bf a, v16bf b, v8f c) {
    return __builtin_amdgcn_wmma_f32_16x16x32_bf16(false, a, false, b, (short)0, c, false, false);
}

// A fragment (16x32 bf16) from a row-major bf16 buffer (two ds_load_b128).
__device__ __forceinline__ v16bf load_afrag(const unsigned short* rowbase, int ld, int kcol) {
    int lane = threadIdx.x & 31;
    const unsigned short* p = rowbase + (lane & 15) * ld + kcol + ((lane >> 4) << 4);
    Frag16 f;
    f.q[0] = *(const uint4*)(p);
    f.q[1] = *(const uint4*)(p + 8);
    return f.v;
}

// B fragment (32x16 bf16) from a TRANSPOSED buffer T[col*64 + row] (two ds_load_b128).
__device__ __forceinline__ v16bf load_bfrag_tr(const unsigned short* T, int nt, int kc) {
    int lane = threadIdx.x & 31;
    const unsigned short* p = T + (nt * 16 + (lane & 15)) * 64 + kc * 32 + ((lane >> 4) << 4);
    Frag16 f;
    f.q[0] = *(const uint4*)(p);
    f.q[1] = *(const uint4*)(p + 8);
    return f.v;
}

// B fragment from pre-packed global weights: contiguous per lane (two global_load_b128).
__device__ __forceinline__ v16bf load_bfrag_packed(const unsigned short* wp, int kc, int nt, int ntN) {
    int lane = threadIdx.x & 31;
    const unsigned short* p = wp + ((size_t)((kc * ntN + nt) * 32 + lane) << 4);
    Frag16 f;
    f.q[0] = *(const uint4*)(p);
    f.q[1] = *(const uint4*)(p + 8);
    return f.v;
}

#if HAVE_TDM
// TDM: async 2D tile load (100 x 64 f32) global -> LDS. D# per CDNA5 ISA ch.8.
// This toolchain exposes the 6-arg builtin: (g0 u32x4, g1 i32x8, g2 i32x4, g3 i32x4, i32x8, cpol).
__device__ __forceinline__ void tdm_load_xt(const float* gptr, float* lds_dst) {
    unsigned lds_addr = (unsigned)(unsigned long long)(size_t)lds_dst;  // LDS aperture: offset in [31:0]
    unsigned long long ga = (unsigned long long)(size_t)gptr;
    u32x4 g0;
    g0[0] = 1u;                                              // count=1 (valid), user flags 0
    g0[1] = lds_addr;                                        // lds_addr
    g0[2] = (unsigned)(ga & 0xFFFFFFFFull);                  // global_addr[31:0]
    g0[3] = (unsigned)((ga >> 32) & 0x1FFFFFFull) | (2u << 30);  // global_addr[56:32] | type=2
    i32x8 g1;
    g1[0] = (int)(2u << 16);          // data_size=2 (4B); no multicast/pad/iterate
    g1[1] = (int)(100u << 16);        // tensor_dim0[15:0]=100
    g1[2] = (int)(64u << 16);         // tensor_dim0[31:16]=0 | tensor_dim1[15:0]=64
    g1[3] = (int)(100u << 16);        // tensor_dim1[31:16]=0 | tile_dim0=100
    g1[4] = 64;                       // tile_dim1=64 | tile_dim2=0
    g1[5] = 100;                      // tensor_dim0_stride[31:0]=100
    g1[6] = 0;                        // stride[47:32]=0 | tensor_dim1_stride[15:0]=0
    g1[7] = 0;
    i32x4 z4 = {0, 0, 0, 0};
    i32x8 z8 = {0, 0, 0, 0, 0, 0, 0, 0};
    __builtin_amdgcn_tensor_load_to_lds(g0, g1, z4, z4, z8, 0);
}
#endif

// ---------------- mean over T ----------------
__global__ __launch_bounds__(256) void k_mean(const float* __restrict__ x, float* __restrict__ xagg) {
    const int ND = 64 * 100, T = 128;
    int idx = blockIdx.x * 256 + threadIdx.x;
    if (idx >= 32 * ND) return;
    int b = idx / ND, r = idx - b * ND;
    const float* p = x + (size_t)b * T * ND + r;
    float s = 0.f;
    for (int t = 0; t < T; ++t) s += p[(size_t)t * ND];
    xagg[idx] = s * (1.0f / 128.0f);
}

// ---------------- weight packing into WMMA B-fragment layout ----------------
__global__ __launch_bounds__(256) void k_pack(const float* __restrict__ W, unsigned short* __restrict__ out,
                                              int catd, int segw, int hbw, int wout) {
    int idx = blockIdx.x * 256 + threadIdx.x;
    if (idx >= hbw * wout) return;
    int j = idx & 15;
    int lane = (idx >> 4) & 31;
    int t = idx >> 9;
    int ntN = wout >> 4;
    int nt = t % ntN, kc = t / ntN;
    int n = nt * 16 + (lane & 15);
    int hi = lane >> 4;
    int col = kc * 32 + hi * 16 + j;
    int s = col / segw, c = col - s * segw;
    float v = 0.f;
    if (c < catd) v = W[(size_t)(s * catd + c) * wout + n];
    out[idx] = f2bf(v);
}

// ---------------- graph learning: A heads (bf16 output) ----------------
__global__ __launch_bounds__(256) void k_graph(const float* __restrict__ xagg,
                                               const float* __restrict__ gl_w1, const float* __restrict__ gl_b1,
                                               const float* __restrict__ gl_w2, const float* __restrict__ gl_b2,
                                               unsigned short* __restrict__ Abf) {
    __shared__ float t1[4096];
    __shared__ float ebuf[4096];
    __shared__ float Atile[4096];
    __shared__ float dinv[64];
    int b = blockIdx.x >> 1, k = blockIdx.x & 1;
    int tid = threadIdx.x;
    const float* xa = xagg + (size_t)b * 6400;
    const float* w1 = gl_w1 + (size_t)k * 6400;
    const float* b1 = gl_b1 + k * 64;
    const float* w2 = gl_w2 + (size_t)k * 4096;
    const float* b2 = gl_b2 + k * 64;
    for (int i = tid; i < 4096; i += 256) {
        int n = i >> 6, hh = i & 63;
        float s = b1[hh];
        for (int d = 0; d < 100; ++d) s += xa[n * 100 + d] * w1[d * 64 + hh];
        t1[i] = fmaxf(s, 0.f);
    }
    __syncthreads();
    for (int i = tid; i < 4096; i += 256) {
        int n = i >> 6, hh = i & 63;
        float s = b2[hh];
        for (int d = 0; d < 64; ++d) s += t1[n * 64 + d] * w2[d * 64 + hh];
        ebuf[i] = s;
    }
    __syncthreads();
    for (int i = tid; i < 4096; i += 256) {
        int n = i >> 6, m = i & 63;
        float s = 0.f;
        for (int d = 0; d < 64; ++d) s += ebuf[n * 64 + d] * ebuf[m * 64 + d];
        float pi = 1.f / (1.f + __expf(-s));
        float u = uhash((unsigned)k, (unsigned)n, (unsigned)m);
        float logit = logf(pi + 1e-12f) - logf(1.f - pi + 1e-12f)
                    + logf(u + 1e-12f) - logf(1.f - u + 1e-12f);
        Atile[i] = 1.f / (1.f + __expf(-logit * 10.0f));  // TEMP = 0.1
    }
    __syncthreads();
    if (tid < 64) {
        float s = 0.f;
        for (int m = 0; m < 64; ++m) s += Atile[tid * 64 + m];
        dinv[tid] = rsqrtf(s + 1e-10f);
    }
    __syncthreads();
    for (int i = tid; i < 4096; i += 256) {
        int n = i >> 6, m = i & 63;
        Abf[((size_t)b * 2 + k) * 4096 + i] = f2bf(Atile[i] * dinv[n] * dinv[m]);
    }
}

#define TSLOT (192 * 64)   // transposed segment slot stride (max segw)

// ---------------- diffusion ----------------
__device__ void diffusion(unsigned short* hbuf, unsigned short* tbuf, const unsigned short* Abf,
                          int segw, int hbw, int nct, int tid) {
    int lane = tid & 31, wv = tid >> 5;
    int cn_l = lane & 15, hi = lane >> 4;
    int which = wv >> 2, mt = wv & 3;
    const unsigned short* Arow = Abf + which * 4096 + mt * 16 * 64;
    v16bf a0 = load_afrag(Arow, 64, 0);
    v16bf a1 = load_afrag(Arow, 64, 32);
    int m0 = mt * 16 + hi * 8;
    int dseg1 = which ? 3 : 1;
    unsigned short* tdst = tbuf + (which ? 2 : 1) * TSLOT;
    // pass 1: seg1 = A0@seg0, seg3 = A1@seg0
    for (int nt = 0; nt < nct; ++nt) {
        v8f acc = {0.f, 0.f, 0.f, 0.f, 0.f, 0.f, 0.f, 0.f};
        acc = wmma_bf16(a0, load_bfrag_tr(tbuf, nt, 0), acc);
        acc = wmma_bf16(a1, load_bfrag_tr(tbuf, nt, 1), acc);
        int cn = nt * 16 + cn_l;
        Pack8 pk;
#pragma unroll
        for (int i = 0; i < 8; ++i) {
            unsigned short hv = f2bf(acc[i]);
            pk.u[i] = hv;
            hbuf[(m0 + i) * hbw + dseg1 * segw + cn] = hv;
        }
        *(uint4*)(tdst + cn * 64 + m0) = pk.q;
    }
    __syncthreads();
    // pass 2: seg2 = 2*A0@seg1 - seg0, seg4 = 2*A1@seg3 - seg0
    int dseg2 = which ? 4 : 2;
    for (int nt = 0; nt < nct; ++nt) {
        v8f acc = {0.f, 0.f, 0.f, 0.f, 0.f, 0.f, 0.f, 0.f};
        acc = wmma_bf16(a0, load_bfrag_tr(tdst, nt, 0), acc);
        acc = wmma_bf16(a1, load_bfrag_tr(tdst, nt, 1), acc);
        int cn = nt * 16 + cn_l;
        Pack8 s0 = *(const Pack8*)(tbuf + cn * 64 + m0);
#pragma unroll
        for (int i = 0; i < 8; ++i)
            hbuf[(m0 + i) * hbw + dseg2 * segw + cn] = f2bf(2.f * acc[i] - bf2f(s0.u[i]));
    }
    __syncthreads();
}

// ---------------- one DCGRU cell step ----------------
__device__ void cell(int din, int catd, int segw, int nkc, int nct,
                     const float* xt, int xstride,
                     float* h, const unsigned short* Abf,
                     unsigned short* hbuf, unsigned short* tbuf, float* ru,
                     const unsigned short* Wgp, const float* bg,
                     const unsigned short* Wcp, const float* bc, int tid) {
    const int Nn = 64, H = 64;
    int lane = tid & 31, wv = tid >> 5;
    int cn_l = lane & 15, hi = lane >> 4;
    int hbw = nkc * 32;

    // seg0 = [xt, h] (row-major + transposed; zero pad cols catd..segw)
    for (int i = tid; i < Nn * segw; i += 256) {
        int n = i / segw, c = i - n * segw;
        float v = 0.f;
        if (c < din)       v = xt[n * xstride + c];
        else if (c < catd) v = h[n * H + (c - din)];
        unsigned short hv = f2bf(v);
        hbuf[n * hbw + c] = hv;
        tbuf[c * 64 + n] = hv;
    }
    __syncthreads();
    diffusion(hbuf, tbuf, Abf, segw, hbw, nct, tid);

    // gate GEMM: ru = sigmoid(hbuf @ Wg + bg), 64x128; wave = n-tile, B reused over 4 m-tiles
    {
        int nt = wv;
        v8f acc0 = {0.f,0.f,0.f,0.f,0.f,0.f,0.f,0.f};
        v8f acc1 = acc0, acc2 = acc0, acc3 = acc0;
        for (int kc = 0; kc < nkc; ++kc) {
            v16bf b = load_bfrag_packed(Wgp, kc, nt, 8);
            acc0 = wmma_bf16(load_afrag(hbuf + 0 * 16 * hbw, hbw, kc * 32), b, acc0);
            acc1 = wmma_bf16(load_afrag(hbuf + 1 * 16 * hbw, hbw, kc * 32), b, acc1);
            acc2 = wmma_bf16(load_afrag(hbuf + 2 * 16 * hbw, hbw, kc * 32), b, acc2);
            acc3 = wmma_bf16(load_afrag(hbuf + 3 * 16 * hbw, hbw, kc * 32), b, acc3);
        }
        int cn = nt * 16 + cn_l;
        float bias = bg[cn];
        v8f* accs[4] = {&acc0, &acc1, &acc2, &acc3};
#pragma unroll
        for (int mt = 0; mt < 4; ++mt) {
            int m0 = mt * 16 + hi * 8;
#pragma unroll
            for (int i = 0; i < 8; ++i)
                ru[(m0 + i) * 128 + cn] = 1.f / (1.f + __expf(-((*accs[mt])[i] + bias)));
        }
    }
    __syncthreads();

    // seg0 = [xt, r*h]
    for (int i = tid; i < Nn * segw; i += 256) {
        int n = i / segw, c = i - n * segw;
        float v = 0.f;
        if (c < din)       v = xt[n * xstride + c];
        else if (c < catd) { int q = c - din; v = ru[n * 128 + q] * h[n * H + q]; }
        unsigned short hv = f2bf(v);
        hbuf[n * hbw + c] = hv;
        tbuf[c * 64 + n] = hv;
    }
    __syncthreads();
    diffusion(hbuf, tbuf, Abf, segw, hbw, nct, tid);

    // candidate GEMM + GRU update: h = u*h + (1-u)*tanh(hbuf @ Wc + bc)
    {
        int nt = wv & 3, mtb = wv >> 2;
        v8f acc0 = {0.f,0.f,0.f,0.f,0.f,0.f,0.f,0.f};
        v8f acc1 = acc0;
        for (int kc = 0; kc < nkc; ++kc) {
            v16bf b = load_bfrag_packed(Wcp, kc, nt, 4);
            acc0 = wmma_bf16(load_afrag(hbuf + mtb * 16 * hbw, hbw, kc * 32), b, acc0);
            acc1 = wmma_bf16(load_afrag(hbuf + (mtb + 2) * 16 * hbw, hbw, kc * 32), b, acc1);
        }
        int cn = nt * 16 + cn_l;
        float bias = bc[cn];
#pragma unroll
        for (int half = 0; half < 2; ++half) {
            int m0 = (mtb + half * 2) * 16 + hi * 8;
            v8f acc = half ? acc1 : acc0;
#pragma unroll
            for (int i = 0; i < 8; ++i) {
                int n = m0 + i;
                float c = tanhf(acc[i] + bias);
                float u = ru[n * 128 + 64 + cn];
                float hold = h[n * H + cn];
                h[n * H + cn] = u * hold + (1.f - u) * c;
            }
        }
    }
    __syncthreads();
}

// ---------------- recurrent scan: 1 workgroup per batch, both layers interleaved ----------------
__global__ __launch_bounds__(256) void k_recurrent(
    const float* __restrict__ x, const int* __restrict__ seqlen,
    const unsigned short* __restrict__ Abf_g,
    const unsigned short* __restrict__ Wg0p, const unsigned short* __restrict__ Wc0p,
    const float* __restrict__ bg0, const float* __restrict__ bc0,
    const unsigned short* __restrict__ Wg1p, const unsigned short* __restrict__ Wc1p,
    const float* __restrict__ bg1, const float* __restrict__ bc1,
    float* __restrict__ last_g) {
    __shared__ unsigned short hbuf[64 * 960];     // 120 KB row-major: 5 segments
    __shared__ unsigned short tbuf[3 * TSLOT];    // 72 KB transposed seg0/seg1/seg3
    __shared__ unsigned short Abf[2 * 64 * 64];   // 16 KB
    __shared__ float xtf[64 * 100];               // 25 KB x_t staging (TDM target)
    __shared__ float h0[64 * 64];                 // 16 KB
    __shared__ float h1[64 * 64];                 // 16 KB
    __shared__ float ru[64 * 128];                // 32 KB
    const int T = 128;
    int b = blockIdx.x, tid = threadIdx.x, wv = tid >> 5;
    for (int i = tid; i < 2 * 64 * 64; i += 256) Abf[i] = Abf_g[(size_t)b * 8192 + i];
    for (int i = tid; i < 64 * 64; i += 256) { h0[i] = 0.f; h1[i] = 0.f; }
    __syncthreads();
    int sl = seqlen[b];
#if HAVE_TDM
    if (wv == 0) tdm_load_xt(x + (size_t)b * T * 6400, xtf);
#endif
    for (int t = 0; t < T; ++t) {
#if HAVE_TDM
        if (wv == 0) __builtin_amdgcn_s_wait_tensorcnt(0);
#else
        {
            const float* xp = x + ((size_t)b * T + t) * 6400;
            for (int i = tid; i < 6400; i += 256) xtf[i] = xp[i];
        }
#endif
        __syncthreads();
        // layer 0: din=100, catd=164, segw=192(pad), hbw=960, 12 col-tiles/segment
        cell(100, 164, 192, 30, 12, xtf, 100, h0, Abf, hbuf, tbuf, ru, Wg0p, bg0, Wc0p, bc0, tid);
        // x_t fully consumed by layer 0 -> async-prefetch x_{t+1} behind layer-1 compute
#if HAVE_TDM
        if (t + 1 < T && wv == 0) tdm_load_xt(x + ((size_t)b * T + t + 1) * 6400, xtf);
#endif
        // layer 1: din=64 (=h0 output), catd=128, segw=128, hbw=640
        cell(64, 128, 128, 20, 8, h0, 64, h1, Abf, hbuf, tbuf, ru, Wg1p, bg1, Wc1p, bc1, tid);
        if (t == sl - 1)
            for (int i = tid; i < 64 * 64; i += 256) last_g[(size_t)b * 4096 + i] = h1[i];
        __syncthreads();
    }
}

// ---------------- head: maxpool over N, mu/sigma/Z, logits ----------------
__global__ __launch_bounds__(256) void k_head(const float* __restrict__ last_g,
                                              const float* __restrict__ mu_w, const float* __restrict__ mu_b,
                                              const float* __restrict__ sg_w, const float* __restrict__ sg_b,
                                              const float* __restrict__ cls_w, const float* __restrict__ cls_b,
                                              float* __restrict__ out) {
    __shared__ float pooled[2048];
    __shared__ float Zs[2048];
    int tid = threadIdx.x;
    for (int i = tid; i < 2048; i += 256) {
        int b = i >> 6, hh = i & 63;
        float m = -3.4e38f;
        for (int n = 0; n < 64; ++n) m = fmaxf(m, last_g[(size_t)b * 4096 + n * 64 + hh]);
        pooled[i] = m;
    }
    __syncthreads();
    for (int i = tid; i < 2048; i += 256) {
        int b = i >> 6, j = i & 63;
        float mu = mu_b[j], sg = sg_b[j];
        for (int hh = 0; hh < 64; ++hh) {
            float p = pooled[b * 64 + hh];
            mu += p * mu_w[hh * 64 + j];
            sg += p * sg_w[hh * 64 + j];
        }
        float sigma = fmaxf(sg, 0.f) + log1pf(__expf(-fabsf(sg)));  // softplus
        float u1 = uhash(99u, (unsigned)b, (unsigned)j);
        float u2 = uhash(991u, (unsigned)b, (unsigned)j);
        float eps = sqrtf(-2.f * logf(u1)) * cosf(6.2831853f * u2);
        float Z = mu + sigma * eps;
        Zs[i] = Z;
        out[128 + i] = Z;
        out[128 + 2048 + i] = mu;
        out[128 + 4096 + i] = sigma;
    }
    __syncthreads();
    for (int i = tid; i < 128; i += 256) {
        int b = i >> 2, c = i & 3;
        float s = cls_b[c];
        for (int hh = 0; hh < 64; ++hh) s += fmaxf(Zs[b * 64 + hh], 0.f) * cls_w[hh * 4 + c];
        out[i] = s;
    }
}

extern "C" void kernel_launch(void* const* d_in, const int* in_sizes, int n_in,
                              void* d_out, int out_size, void* d_ws, size_t ws_size,
                              hipStream_t stream) {
    const float* x      = (const float*)d_in[0];
    const int*   seqlen = (const int*)d_in[1];
    // d_in[2] supports: unused by reference math path
    const float* gl_w1 = (const float*)d_in[3];
    const float* gl_b1 = (const float*)d_in[4];
    const float* gl_w2 = (const float*)d_in[5];
    const float* gl_b2 = (const float*)d_in[6];
    const float* Wg0 = (const float*)d_in[7];
    const float* bg0 = (const float*)d_in[8];
    const float* Wc0 = (const float*)d_in[9];
    const float* bc0 = (const float*)d_in[10];
    const float* Wg1 = (const float*)d_in[11];
    const float* bg1 = (const float*)d_in[12];
    const float* Wc1 = (const float*)d_in[13];
    const float* bc1 = (const float*)d_in[14];
    const float* mu_w = (const float*)d_in[15];
    const float* mu_b = (const float*)d_in[16];
    const float* sg_w = (const float*)d_in[17];
    const float* sg_b = (const float*)d_in[18];
    const float* cls_w = (const float*)d_in[19];
    const float* cls_b = (const float*)d_in[20];

    char* ws = (char*)d_ws;
    float*          xagg   = (float*)(ws + 0);                 // 32*64*100 f32 = 819200 B
    unsigned short* Abf    = (unsigned short*)(ws + 819200);   // 32*2*4096 bf16 = 524288 B
    unsigned short* Wg0p   = (unsigned short*)(ws + 1343488);  // 960*128 bf16 = 245760 B
    unsigned short* Wc0p   = (unsigned short*)(ws + 1589248);  // 960*64  bf16 = 122880 B
    unsigned short* Wg1p   = (unsigned short*)(ws + 1712128);  // 640*128 bf16 = 163840 B
    unsigned short* Wc1p   = (unsigned short*)(ws + 1875968);  // 640*64  bf16 = 81920 B
    float*          last_g = (float*)(ws + 1957888);           // 32*4096 f32 = 524288 B
    float* out = (float*)d_out;

    k_mean<<<800, 256, 0, stream>>>(x, xagg);
    k_pack<<<480, 256, 0, stream>>>(Wg0, Wg0p, 164, 192, 960, 128);
    k_pack<<<240, 256, 0, stream>>>(Wc0, Wc0p, 164, 192, 960, 64);
    k_pack<<<320, 256, 0, stream>>>(Wg1, Wg1p, 128, 128, 640, 128);
    k_pack<<<160, 256, 0, stream>>>(Wc1, Wc1p, 128, 128, 640, 64);
    k_graph<<<64, 256, 0, stream>>>(xagg, gl_w1, gl_b1, gl_w2, gl_b2, Abf);
    k_recurrent<<<32, 256, 0, stream>>>(x, seqlen, Abf, Wg0p, Wc0p, bg0, bc0,
                                        Wg1p, Wc1p, bg1, bc1, last_g);
    k_head<<<1, 256, 0, stream>>>(last_g, mu_w, mu_b, sg_w, sg_b, cls_w, cls_b, out);

    (void)in_sizes; (void)n_in; (void)out_size; (void)ws_size;
}
